// NceNet_65094524338890
// MI455X (gfx1250) — matrix operand order repaired and checked by reading, not compiled, not run on hardware
//
#include <hip/hip_runtime.h>
#include <hip/hip_bf16.h>

// ---------------------------------------------------------------------------
// NCE loss, MI455X (gfx1250, wave32, WMMA).
// loss = mean_m -( logsig(e_u.e_v) + logsig(-(sum_k e_neg_k).e_v) )
// Memory-bound gather kernel (~184 MB @ 23.3 TB/s ~ 8us); dots done with
// v_wmma_f32_16x16x32_bf16 on a 16-row tile per wave (diagonal of 16x16).
// ---------------------------------------------------------------------------

typedef __attribute__((ext_vector_type(16))) __bf16    v16bf;
typedef __attribute__((ext_vector_type(8)))  float     v8f;
typedef __attribute__((ext_vector_type(8)))  unsigned  v8u;

#define EDIM  128
#define KNEG  20
#define LSTR  66          // LDS row stride in uints (padded: 66%64=2 -> no bank conflict)

static __device__ __forceinline__ unsigned pack_bf16(float a, float b) {
    // round-to-nearest-even f32 -> bf16, packed {lo=a, hi=b}
    unsigned ua = __float_as_uint(a);
    unsigned ub = __float_as_uint(b);
    ua += 0x7FFFu + ((ua >> 16) & 1u);
    ub += 0x7FFFu + ((ub >> 16) & 1u);
    return (ua >> 16) | (ub & 0xFFFF0000u);
}

static __device__ __forceinline__ float log_sigmoid(float x) {
    return fminf(x, 0.0f) - log1pf(expf(-fabsf(x)));
}

__global__ __launch_bounds__(32)
void nce_tile_kernel(const int*   __restrict__ u,
                     const int*   __restrict__ u_neg,
                     const int*   __restrict__ v,
                     const float* __restrict__ Eu,
                     const float* __restrict__ Ev,
                     float*       __restrict__ partial)
{
    __shared__ unsigned sAu[16 * LSTR];   // e_u tile, bf16-packed (row-major K)
    __shared__ unsigned sAn[16 * LSTR];   // sum of 20 negatives, bf16-packed
    __shared__ unsigned sBv[16 * LSTR];   // e_v tile (columns of B)

    const int lane = threadIdx.x;               // 0..31
    const int r0   = blockIdx.x * 16;           // 16 rows per wave/block
    const float4* Eu4 = reinterpret_cast<const float4*>(Eu);
    const float4* Ev4 = reinterpret_cast<const float4*>(Ev);

    // ---- gather phase: each lane owns elements [4*lane .. 4*lane+3] of a row
    for (int i = 0; i < 16; ++i) {
        const int row = r0 + i;

        const float4 xv = Ev4[(size_t)v[row] * (EDIM / 4) + lane];
        sBv[i * LSTR + 2 * lane + 0] = pack_bf16(xv.x, xv.y);
        sBv[i * LSTR + 2 * lane + 1] = pack_bf16(xv.z, xv.w);

        const float4 xu = Eu4[(size_t)u[row] * (EDIM / 4) + lane];
        sAu[i * LSTR + 2 * lane + 0] = pack_bf16(xu.x, xu.y);
        sAu[i * LSTR + 2 * lane + 1] = pack_bf16(xu.z, xu.w);

        float4 acc = make_float4(0.f, 0.f, 0.f, 0.f);
        const int* np = u_neg + (size_t)row * KNEG;
#pragma unroll
        for (int k = 0; k < KNEG; ++k) {        // 20 loads in flight (LOADcnt)
            const float4 t = Eu4[(size_t)np[k] * (EDIM / 4) + lane];
            acc.x += t.x; acc.y += t.y; acc.z += t.z; acc.w += t.w;
        }
        sAn[i * LSTR + 2 * lane + 0] = pack_bf16(acc.x, acc.y);
        sAn[i * LSTR + 2 * lane + 1] = pack_bf16(acc.z, acc.w);
    }
    __syncthreads();  // single-wave WG: barrier is S_NOP, just orders LDS (DScnt)

    // ---- WMMA operand fetch per CDNA5 ISA VGPR layouts (uint = 2 bf16) ----
    // A 16x32 (16-bit): lane L, M=L&15: V0-3 -> K=(L<16?0:8)+{2j,2j+1}, V4-7 -> K=16+...
    // B 32x16 (16-bit): lane L, N=L&15:   V0-3 -> K=(L<16?0:8)+{2j,2j+1}, V4-7 -> K=16+...
    const int r   = lane & 15;
    const int aKb = (lane < 16) ? 0 : 4;   // A K-block offset (uint units)
    const int bKb = (lane < 16) ? 0 : 4;   // B K-block offset (uint units)

    v8f c_uu = {};   // D = e_u  x e_v^T  (need diagonal)
    v8f c_nv = {};   // D = nsum x e_v^T  (need diagonal)
#pragma unroll
    for (int c = 0; c < 4; ++c) {          // K = 128 in 4 chunks of 32
        const int base = c * 16;           // 32 bf16 = 16 uints per chunk
        v8u ra, rn, rb;
#pragma unroll
        for (int j = 0; j < 8; ++j) {
            const int aoff = base + ((j < 4) ? (aKb + j) : (8 + aKb + (j - 4)));
            const int boff = base + ((j < 4) ? (bKb + j) : (8 + bKb + (j - 4)));
            ra[j] = sAu[r * LSTR + aoff];
            rn[j] = sAn[r * LSTR + aoff];
            rb[j] = sBv[r * LSTR + boff];
        }
        const v16bf A = __builtin_bit_cast(v16bf, ra);
        const v16bf N = __builtin_bit_cast(v16bf, rn);
        const v16bf B = __builtin_bit_cast(v16bf, rb);
        c_uu = __builtin_amdgcn_wmma_f32_16x16x32_bf16(false, A, false, B,
                                                       (short)0, c_uu, false, false);
        c_nv = __builtin_amdgcn_wmma_f32_16x16x32_bf16(false, N, false, B,
                                                       (short)0, c_nv, false, false);
    }

    // ---- diagonal of 16x16 f32 D: (i,i) lives in lane i (i<8, VGPR i) or
    //      lane i+16 (i>=8, VGPR i-8)
    int g = -1;
    if (lane < 8)        g = lane;          // rows 0..7
    else if (lane >= 24) g = lane - 24;     // rows 8..15
    float duu = 0.f, dnv = 0.f;
#pragma unroll
    for (int j = 0; j < 8; ++j)
        if (j == g) { duu = c_uu[j]; dnv = c_nv[j]; }

    float contrib = 0.f;
    if (g >= 0)
        contrib = -(log_sigmoid(duu) + log_sigmoid(-dnv));

    // deterministic wave32 reduction
#pragma unroll
    for (int off = 16; off > 0; off >>= 1)
        contrib += __shfl_xor(contrib, off, 32);
    if (lane == 0)
        partial[blockIdx.x] = contrib;
}

__global__ __launch_bounds__(256)
void nce_reduce_kernel(const float* __restrict__ partial, float* __restrict__ out,
                       int n, float scale)
{
    __shared__ float s[256];
    float acc = 0.f;
    for (int i = threadIdx.x; i < n; i += 256) acc += partial[i];
    s[threadIdx.x] = acc;
    __syncthreads();
    for (int st = 128; st > 0; st >>= 1) {
        if ((int)threadIdx.x < st) s[threadIdx.x] += s[threadIdx.x + st];
        __syncthreads();
    }
    if (threadIdx.x == 0) out[0] = s[0] * scale;
}

extern "C" void kernel_launch(void* const* d_in, const int* in_sizes, int n_in,
                              void* d_out, int out_size, void* d_ws, size_t ws_size,
                              hipStream_t stream)
{
    const int*   u     = (const int*)d_in[0];
    const int*   u_neg = (const int*)d_in[1];
    const int*   v     = (const int*)d_in[2];
    const float* Eu    = (const float*)d_in[3];
    const float* Ev    = (const float*)d_in[4];
    float* out     = (float*)d_out;
    float* partial = (float*)d_ws;

    const int M      = in_sizes[0];     // 16384
    const int ntiles = M / 16;          // 1024 waves

    nce_tile_kernel<<<ntiles, 32, 0, stream>>>(u, u_neg, v, Eu, Ev, partial);
    nce_reduce_kernel<<<1, 256, 0, stream>>>(partial, out, ntiles, 1.0f / (float)M);
}
